// GraphTransformerEncoder_17308718203035
// MI455X (gfx1250) — compile-verified, matrix operand order checked by reading, not run on hardware
//
#include <hip/hip_runtime.h>

// Problem constants (match reference)
#define G_    128
#define NPER  256
#define NTOT  (G_ * NPER)      // 32768 nodes
#define D_    256
#define H_    8
#define DH_   32
#define L_    3
#define INDIM 128

typedef __attribute__((ext_vector_type(16))) _Float16 v16h;
typedef __attribute__((ext_vector_type(8)))  _Float16 v8h;
typedef __attribute__((ext_vector_type(8)))  float    v8f;

// ---------------------------------------------------------------------------
// CDNA5 WMMA: D(16x16 f32) = A(16x32 f16) * B(32x16 f16) + C
// ---------------------------------------------------------------------------
static __device__ __forceinline__ v8f wmma16x16x32(v16h a, v16h b, v8f c) {
  return __builtin_amdgcn_wmma_f32_16x16x32_f16(
      /*neg_a=*/false, a, /*neg_b=*/false, b,
      /*c_mod=*/(short)0, c, /*reuse_a=*/false, /*reuse_b=*/false);
}

// Fragment from a row-major f16 row: half i <-> k = (i/8)*16 + kb8 + (i%8),
// kb8 = 8*(lane>=16). Two aligned 16B loads (global or LDS), no conversion.
static __device__ __forceinline__ v16h frag_h(const _Float16* p, int kb8) {
  v8h lo = *(const v8h*)(p + kb8);
  v8h hi = *(const v8h*)(p + kb8 + 16);
  return __builtin_shufflevector(lo, hi, 0, 1, 2, 3, 4, 5, 6, 7,
                                 8, 9, 10, 11, 12, 13, 14, 15);
}

// wave32 cross-lane reductions (xor masks < 16 stay within each 16-lane half)
static __device__ __forceinline__ float redmax16(float v) {
  v = fmaxf(v, __shfl_xor(v, 1, 32));
  v = fmaxf(v, __shfl_xor(v, 2, 32));
  v = fmaxf(v, __shfl_xor(v, 4, 32));
  v = fmaxf(v, __shfl_xor(v, 8, 32));
  return v;
}
static __device__ __forceinline__ float redsum16(float v) {
  v += __shfl_xor(v, 1, 32); v += __shfl_xor(v, 2, 32);
  v += __shfl_xor(v, 4, 32); v += __shfl_xor(v, 8, 32);
  return v;
}
static __device__ __forceinline__ float redsum32(float v) {
  v += __shfl_xor(v, 1, 32);  v += __shfl_xor(v, 2, 32);
  v += __shfl_xor(v, 4, 32);  v += __shfl_xor(v, 8, 32);
  v += __shfl_xor(v, 16, 32);
  return v;
}

// ---------------------------------------------------------------------------
// f32 -> f16 bulk conversion (n must be a multiple of 8; true for all uses)
// ---------------------------------------------------------------------------
__global__ __launch_bounds__(256)
void cvt16(const float* __restrict__ src, _Float16* __restrict__ dst, int n) {
  int i = (blockIdx.x * 256 + threadIdx.x) * 8;
  if (i >= n) return;
  float4 u0 = *(const float4*)(src + i);
  float4 u1 = *(const float4*)(src + i + 4);
  v8h d;
  d[0] = (_Float16)u0.x; d[1] = (_Float16)u0.y;
  d[2] = (_Float16)u0.z; d[3] = (_Float16)u0.w;
  d[4] = (_Float16)u1.x; d[5] = (_Float16)u1.y;
  d[6] = (_Float16)u1.z; d[7] = (_Float16)u1.w;
  *(v8h*)(dst + i) = d;
}

// ---------------------------------------------------------------------------
// GEMM: C[M,N] = act(A[M,K] @ W[N,K]^T + bias[N]), A and W in f16.
// One wave computes a 32x64 tile: 2 A-frags x 4 B-frags = 8 WMMAs per K-step.
// Optional f32 and/or f16 outputs (either pointer may be null).
// ---------------------------------------------------------------------------
__global__ __launch_bounds__(128)
void gemm32x64(const _Float16* __restrict__ A, const _Float16* __restrict__ W,
               const float* __restrict__ bias, float* __restrict__ C32,
               _Float16* __restrict__ C16, int M, int N, int K, int relu) {
  int wid  = (blockIdx.x * 128 + threadIdx.x) >> 5;
  int lane = threadIdx.x & 31;
  int tiles_n = N >> 6;
  int total   = (M >> 5) * tiles_n;
  if (wid >= total) return;                 // whole-wave exit, EXEC stays full
  int tm  = wid / tiles_n;
  int tn  = wid - tm * tiles_n;
  int l16 = lane & 15;
  int kb8 = (lane >> 4) << 3;               // 0 or 8

  const _Float16* arow0 = A + (size_t)(tm * 32 + l16) * K;
  const _Float16* arow1 = A + (size_t)(tm * 32 + 16 + l16) * K;
  const _Float16* wrow  = W + (size_t)(tn * 64 + l16) * K;
  const size_t KS16 = (size_t)16 * K;

  v8f acc[8];
#pragma unroll
  for (int i = 0; i < 8; ++i) acc[i] = (v8f){};

  for (int k0 = 0; k0 < K; k0 += 32) {
    v16h a0 = frag_h(arow0 + k0, kb8);
    v16h a1 = frag_h(arow1 + k0, kb8);
#pragma unroll
    for (int r = 0; r < 4; ++r) {
      v16h b = frag_h(wrow + (size_t)r * KS16 + k0, kb8);
      acc[2 * r]     = wmma16x16x32(a0, b, acc[2 * r]);
      acc[2 * r + 1] = wmma16x16x32(a1, b, acc[2 * r + 1]);
    }
  }

#pragma unroll
  for (int r = 0; r < 4; ++r) {
    int n = tn * 64 + r * 16 + l16;
    float bs = bias[n];
#pragma unroll
    for (int p = 0; p < 2; ++p) {
      int mrow = tm * 32 + p * 16 + kb8;    // C layout: vgpr j -> row mrow+j
#pragma unroll
      for (int j = 0; j < 8; ++j) {
        float v = acc[2 * r + p][j] + bs;
        if (relu) v = fmaxf(v, 0.0f);
        if (C32) C32[(size_t)(mrow + j) * N + n] = v;
        if (C16) C16[(size_t)(mrow + j) * N + n] = (_Float16)v;
      }
    }
  }
}

// ---------------------------------------------------------------------------
// Fused flash attention, one block per (graph, head). qkv is f16 [N, 3D].
// K staged into LDS with GLOBAL_LOAD_ASYNC_TO_LDS_B128 (ASYNCcnt path, no
// VGPR round-trip); V staged TRANSPOSED (Vt[d][key]) via the explicit path so
// the P*V B-fragments are contiguous ds_load_b128 pairs. 1/sqrt(DH) applied
// to the f32 scores. Output written f16 (feeds the out-projection GEMM).
// ---------------------------------------------------------------------------
#define KSRS 40   // K row stride (halves): 80B, multiple of 16B
#define VTRS 264  // Vt row stride (halves): 528B, multiple of 16B
#define PRS  40   // P row stride (halves)

__global__ __launch_bounds__(128)
void attn_kernel(const _Float16* __restrict__ qkv, _Float16* __restrict__ o) {
  __shared__ _Float16 Ks[NPER * KSRS];       // 20 KB
  __shared__ _Float16 Vt[DH_ * VTRS];        // 16.5 KB
  __shared__ _Float16 Ps[4 * 16 * PRS];      // 5 KB

  int g    = blockIdx.x >> 3;
  int hh   = blockIdx.x & 7;
  int tid  = threadIdx.x;
  int wave = tid >> 5;
  int lane = tid & 31;
  int l16  = lane & 15;
  int kb8  = (lane >> 4) << 3;               // 0 or 8
  size_t nb = (size_t)g * NPER;

  // Stage K (async global->LDS, 16B per op) and V (transpose via VGPRs).
  for (int c = tid; c < (NPER * DH_) / 8; c += 128) {
    int key = c >> 2;
    int d8  = (c & 3) << 3;
    const _Float16* base = qkv + (nb + key) * (3 * D_) + hh * DH_ + d8;
    // K rows: async DMA into LDS (tracked by ASYNCcnt)
    unsigned ldsa = (unsigned)(unsigned long long)&Ks[key * KSRS + d8];
    asm volatile("global_load_async_to_lds_b128 %0, %1, off"
                 :: "v"(ldsa), "v"(base + D_) : "memory");
    // V columns: load, transpose, scatter to LDS
    v8h vv = *(const v8h*)(base + 2 * D_);
#pragma unroll
    for (int t = 0; t < 8; ++t) Vt[(d8 + t) * VTRS + key] = vv[t];
  }
  asm volatile("s_wait_asynccnt 0" ::: "memory");  // K writes visible in LDS
  __syncthreads();

  const float scale = 0.17677669529663687f;   // 1/sqrt(32)
  _Float16* Pw = Ps + wave * 16 * PRS;

  for (int it = 0; it < 4; ++it) {            // 4 q-blocks per wave
    int qb = wave * 4 + it;
    v16h af = frag_h(qkv + (nb + qb * 16 + l16) * (3 * D_) + hh * DH_, kb8);

    v8f O0 = {}, O1 = {};
    float mrun[8], lrun[8];
#pragma unroll
    for (int j = 0; j < 8; ++j) { mrun[j] = -3.0e38f; lrun[j] = 0.0f; }

    for (int kt = 0; kt < 8; ++kt) {          // 32 keys per iteration
      v16h bk0 = frag_h(Ks + (kt * 32 + l16) * KSRS, kb8);
      v16h bk1 = frag_h(Ks + (kt * 32 + 16 + l16) * KSRS, kb8);
      v8f z = {};
      v8f S0 = wmma16x16x32(af, bk0, z);
      v8f S1 = wmma16x16x32(af, bk1, z);

      // Online softmax on the C layout (row = j + kb8, cols across 16 lanes)
#pragma unroll
      for (int j = 0; j < 8; ++j) {
        float s0 = S0[j] * scale;
        float s1 = S1[j] * scale;
        float c  = redmax16(fmaxf(s0, s1));
        float nm = fmaxf(mrun[j], c);
        float sc = __expf(mrun[j] - nm);
        float p0 = __expf(s0 - nm);
        float p1 = __expf(s1 - nm);
        float rs = redsum16(p0 + p1);
        mrun[j] = nm;
        lrun[j] = lrun[j] * sc + rs;
        O0[j] *= sc;
        O1[j] *= sc;
        int row = j + kb8;
        Pw[row * PRS + l16]      = (_Float16)p0;      // C-layout -> LDS
        Pw[row * PRS + 16 + l16] = (_Float16)p1;
      }

      // P in A-operand layout; V columns contiguous thanks to the transpose.
      // (LDS ops are in-order within a wave: store->load ordering is safe.)
      v16h pa  = frag_h(Pw + l16 * PRS, kb8);
      v16h bv0 = frag_h(Vt + l16 * VTRS + kt * 32, kb8);
      v16h bv1 = frag_h(Vt + (16 + l16) * VTRS + kt * 32, kb8);
      O0 = wmma16x16x32(pa, bv0, O0);
      O1 = wmma16x16x32(pa, bv1, O1);
    }

    // Normalize and write O (16 rows x 32 d-cols) as f16.
#pragma unroll
    for (int j = 0; j < 8; ++j) {
      float inv = 1.0f / lrun[j];
      size_t row = nb + (size_t)qb * 16 + kb8 + j;
      o[row * D_ + hh * DH_ + l16]      = (_Float16)(O0[j] * inv);
      o[row * D_ + hh * DH_ + 16 + l16] = (_Float16)(O1[j] * inv);
    }
  }
}

// ---------------------------------------------------------------------------
// LayerNorm over D=256, optional residual add, optional dual f32+f16 output.
// One wave per row; vectorized 16B accesses.
// ---------------------------------------------------------------------------
template <bool RES, bool DUAL>
__global__ __launch_bounds__(256)
void ln_kernel(const float* __restrict__ x, const float* __restrict__ r,
               const float* __restrict__ g, const float* __restrict__ b,
               float* __restrict__ out32, _Float16* __restrict__ out16,
               int rows) {
  int wid  = (blockIdx.x * 256 + threadIdx.x) >> 5;
  int lane = threadIdx.x & 31;
  if (wid >= rows) return;

  const float* xr = x + (size_t)wid * D_;
  int c0 = lane * 8;
  float v[8];
  {
    const float4* p = (const float4*)(xr + c0);
    float4 u0 = p[0], u1 = p[1];
    v[0] = u0.x; v[1] = u0.y; v[2] = u0.z; v[3] = u0.w;
    v[4] = u1.x; v[5] = u1.y; v[6] = u1.z; v[7] = u1.w;
  }
  if (RES) {
    const float4* q = (const float4*)(r + (size_t)wid * D_ + c0);
    float4 w0 = q[0], w1 = q[1];
    v[0] += w0.x; v[1] += w0.y; v[2] += w0.z; v[3] += w0.w;
    v[4] += w1.x; v[5] += w1.y; v[6] += w1.z; v[7] += w1.w;
  }
  float s = 0.f;
#pragma unroll
  for (int i = 0; i < 8; ++i) s += v[i];
  s = redsum32(s);
  float mu = s * (1.0f / D_);
  float q2 = 0.f;
#pragma unroll
  for (int i = 0; i < 8; ++i) { float d = v[i] - mu; q2 += d * d; }
  q2 = redsum32(q2);
  float inv = rsqrtf(q2 * (1.0f / D_) + 1e-5f);

  float o[8];
#pragma unroll
  for (int i = 0; i < 8; ++i)
    o[i] = (v[i] - mu) * inv * g[c0 + i] + b[c0 + i];

  float4* po = (float4*)(out32 + (size_t)wid * D_ + c0);
  po[0] = make_float4(o[0], o[1], o[2], o[3]);
  po[1] = make_float4(o[4], o[5], o[6], o[7]);
  if (DUAL) {
    v8h hv;
#pragma unroll
    for (int i = 0; i < 8; ++i) hv[i] = (_Float16)o[i];
    *(v8h*)(out16 + (size_t)wid * D_ + c0) = hv;
  }
}

// ---------------------------------------------------------------------------
// Host orchestration
// ---------------------------------------------------------------------------
extern "C" void kernel_launch(void* const* d_in, const int* in_sizes, int n_in,
                              void* d_out, int out_size, void* d_ws,
                              size_t ws_size, hipStream_t stream) {
  (void)in_sizes; (void)n_in; (void)out_size; (void)ws_size;
  const float* x     = (const float*)d_in[0];
  // d_in[1]=edge_index, d_in[2]=batch: unused (graphs are contiguous runs)
  const float* Wp    = (const float*)d_in[3];
  const float* bp    = (const float*)d_in[4];
  const float* in_w  = (const float*)d_in[5];
  const float* in_b  = (const float*)d_in[6];
  const float* out_w = (const float*)d_in[7];
  const float* out_b = (const float*)d_in[8];
  const float* n1g   = (const float*)d_in[9];
  const float* n1b   = (const float*)d_in[10];
  const float* W1    = (const float*)d_in[11];
  const float* b1    = (const float*)d_in[12];
  const float* W2    = (const float*)d_in[13];
  const float* b2    = (const float*)d_in[14];
  const float* n2g   = (const float*)d_in[15];
  const float* n2b   = (const float*)d_in[16];
  const float* lng   = (const float*)d_in[17];
  const float* lnb   = (const float*)d_in[18];

  // ---- workspace carve (all region sizes are multiples of 256B) ----
  char* w = (char*)d_ws;
  float*     h32   = (float*)w;      w += (size_t)NTOT * D_ * 4;      // 32 MB
  float*     tmp32 = (float*)w;      w += (size_t)NTOT * D_ * 4;      // 32 MB
  _Float16*  h16   = (_Float16*)w;   w += (size_t)NTOT * D_ * 2;      // 16 MB
  _Float16*  qkv16 = (_Float16*)w;   w += (size_t)NTOT * 3 * D_ * 2;  // 48 MB
  _Float16*  o16   = (_Float16*)w;   w += (size_t)NTOT * D_ * 2;      // 16 MB
  _Float16*  f116  = (_Float16*)w;   w += (size_t)NTOT * 2 * D_ * 2;  // 32 MB
  _Float16*  x16   = (_Float16*)w;   w += (size_t)NTOT * INDIM * 2;   //  8 MB
  _Float16*  wp16  = (_Float16*)w;   w += (size_t)D_ * INDIM * 2;
  _Float16*  inw16 = (_Float16*)w;   w += (size_t)L_ * 3 * D_ * D_ * 2;
  _Float16*  otw16 = (_Float16*)w;   w += (size_t)L_ * D_ * D_ * 2;
  _Float16*  w116  = (_Float16*)w;   w += (size_t)L_ * 2 * D_ * D_ * 2;
  _Float16*  w216  = (_Float16*)w;   w += (size_t)L_ * D_ * 2 * D_ * 2;

  auto cvt = [&](const float* src, _Float16* dst, int n) {
    cvt16<<<(n / 8 + 255) / 256, 256, 0, stream>>>(src, dst, n);
  };
  auto gemm = [&](const _Float16* A, const _Float16* W, const float* bias,
                  float* C32, _Float16* C16, int M, int N, int K, int relu) {
    int blocks = ((M >> 5) * (N >> 6)) / 4;   // 4 waves per block
    gemm32x64<<<blocks, 128, 0, stream>>>(A, W, bias, C32, C16, M, N, K, relu);
  };

  // One-shot f16 conversion of inputs and weights (tiny vs. GEMM work).
  cvt(x,     x16,   NTOT * INDIM);
  cvt(Wp,    wp16,  D_ * INDIM);
  cvt(in_w,  inw16, L_ * 3 * D_ * D_);
  cvt(out_w, otw16, L_ * D_ * D_);
  cvt(W1,    w116,  L_ * 2 * D_ * D_);
  cvt(W2,    w216,  L_ * D_ * 2 * D_);

  // Input projection: h = x @ Wp^T + bp  (dual f32 + f16 output)
  gemm(x16, wp16, bp, h32, h16, NTOT, D_, INDIM, 0);

  for (int l = 0; l < L_; ++l) {
    // QKV projection (f16-only output; only attention consumes it)
    gemm(h16, inw16 + (size_t)l * 3 * D_ * D_, in_b + (size_t)l * 3 * D_,
         nullptr, qkv16, NTOT, 3 * D_, D_, 0);
    // Fused per-(graph,head) flash attention -> f16
    attn_kernel<<<G_ * H_, 128, 0, stream>>>(qkv16, o16);
    // Output projection (f32; feeds residual)
    gemm(o16, otw16 + (size_t)l * D_ * D_, out_b + (size_t)l * D_,
         tmp32, nullptr, NTOT, D_, D_, 0);
    // h = LN(h + o), dual write
    ln_kernel<true, true><<<NTOT / 8, 256, 0, stream>>>(
        h32, tmp32, n1g + l * D_, n1b + l * D_, h32, h16, NTOT);
    // FFN (hidden kept f16-only; ReLU fused)
    gemm(h16, w116 + (size_t)l * 2 * D_ * D_, b1 + (size_t)l * 2 * D_,
         nullptr, f116, NTOT, 2 * D_, D_, 1);
    gemm(f116, w216 + (size_t)l * D_ * 2 * D_, b2 + (size_t)l * D_,
         tmp32, nullptr, NTOT, D_, 2 * D_, 0);
    // h = LN(h + f), dual write
    ln_kernel<true, true><<<NTOT / 8, 256, 0, stream>>>(
        h32, tmp32, n2g + l * D_, n2b + l * D_, h32, h16, NTOT);
  }

  // Final LayerNorm -> d_out (f32 only)
  ln_kernel<false, false><<<NTOT / 8, 256, 0, stream>>>(
      h32, nullptr, lng, lnb, (float*)d_out, nullptr, NTOT);
}
